// AttentionDecoder_43233140802062
// MI455X (gfx1250) — matrix operand order repaired and checked by reading, not compiled, not run on hardware
//
#include <hip/hip_runtime.h>
#include <hip/hip_bf16.h>

// ---------------------------------------------------------------------------
// AttentionDecoder for MI455X (gfx1250, wave32, WMMA).
//
//  * Dominant cost: logits GEMM (32 x 32000 x 1024) x 128 steps. Wout packed
//    once to bf16 in WMMA-fragment order (65.5 MB, L2-resident); each step
//    streams it exactly once with fully coalesced 32B/lane loads (dual-M-tile
//    waves share the B fragment).
//  * All GEMMs are X @ W^T -> v_wmma_f32_16x16x32_bf16, f32 accumulation.
//  * key_proj hoisted out of the time loop; enc converted to bf16 once.
//  * qproj computed with WMMA too (A = h1 bf16 mirror from previous step).
// ---------------------------------------------------------------------------

#define B_    32
#define S_    256
#define T_    128
#define EENC  512
#define H_    512
#define EEMB  256
#define V_    32000

typedef __attribute__((ext_vector_type(16))) __bf16 v16bf;
typedef __attribute__((ext_vector_type(8)))  __bf16 v8bf;
typedef __attribute__((ext_vector_type(8)))  float  v8f;

__device__ __forceinline__ __bf16 f2bf(float f) {
  union { float f; unsigned u; } v; v.f = f;
  unsigned r = v.u + 0x7FFFu + ((v.u >> 16) & 1u);   // round-to-nearest-even
  unsigned short h = (unsigned short)(r >> 16);
  return __builtin_bit_cast(__bf16, h);
}

__device__ __forceinline__ float sigm(float x) { return 1.0f / (1.0f + __expf(-x)); }

// A fragment (16x32 bf16), ISA 7.12.2: lane m=lane&15, half=lane>>4.
// element e<8  -> k = k0 + 8*half + e          (16B contiguous)
// element e>=8 -> k = k0 + 16 + 8*half + (e-8) (16B contiguous)
__device__ __forceinline__ v16bf load_A_bf16(const __bf16* __restrict__ X, int ldx,
                                             int row0, int k0, int lane) {
  const int m = lane & 15, half = lane >> 4;
  const __bf16* p = X + (size_t)(row0 + m) * ldx + k0 + 8 * half;
  v8bf lo = *(const v8bf*)(p);
  v8bf hi = *(const v8bf*)(p + 16);
  v16bf a;
#pragma unroll
  for (int e = 0; e < 8; ++e) { a[e] = lo[e]; a[e + 8] = hi[e]; }
  return a;
}

// B fragment from weights pre-packed in fragment order:
// chunk (tn, kk) = 1KB contiguous block, [lane][element] -> one 32B load/lane.
__device__ __forceinline__ v16bf load_B_pack(const __bf16* __restrict__ Wp,
                                             size_t chunk, int lane) {
  return *(const v16bf*)(Wp + (chunk * 32 + (size_t)lane) * 16);
}

// ---------------------------------------------------------------------------
// Setup kernels (once per launch)
// ---------------------------------------------------------------------------
__global__ void cvt_bf16_k(const float* __restrict__ src, __bf16* __restrict__ dst, int n) {
  for (int i = blockIdx.x * blockDim.x + threadIdx.x; i < n; i += gridDim.x * blockDim.x)
    dst[i] = f2bf(src[i]);
}

// Pack [A | B] (N x (ca+cb), row-major, f32) into bf16 WMMA-B fragment order:
// dst[(((tn*(K/32)+kk)*32)+lane)*16+e] = W[tn*16+(lane&15)][kk*32+8*(lane>>4)+(e&7)+16*(e>>3)]
__global__ void pack_frag_k(const float* __restrict__ a, int ca,
                            const float* __restrict__ b, int cb,
                            __bf16* __restrict__ dst, int N, int K) {
  const int ksteps = K >> 5;
  const int total = N * K;
  for (int i = blockIdx.x * blockDim.x + threadIdx.x; i < total; i += gridDim.x * blockDim.x) {
    int e    = i & 15;
    int l    = (i >> 4) & 31;
    int rest = i >> 9;
    int kk   = rest % ksteps;
    int tn   = rest / ksteps;
    int row  = tn * 16 + (l & 15);
    int col  = kk * 32 + ((l >> 4) << 3) + (e & 7) + ((e >> 3) << 4);
    float v = (col < ca) ? a[(size_t)row * ca + col] : b[(size_t)row * cb + (col - ca)];
    dst[i] = f2bf(v);
  }
}

__global__ void copy_f32_k(float* __restrict__ dst, const float* __restrict__ src, int n) {
  for (int i = blockIdx.x * blockDim.x + threadIdx.x; i < n; i += gridDim.x * blockDim.x)
    dst[i] = src[i];
}

// key_proj (B*S x H) = enc_bf16 (B*S x EENC) @ Wk^T + bk     [once per launch]
__global__ __launch_bounds__(256)
void gemm_keyproj(const __bf16* __restrict__ encbf, const __bf16* __restrict__ Wkpk,
                  const float* __restrict__ bk, float* __restrict__ out) {
  const int lane = threadIdx.x & 31;
  const int tile = blockIdx.x * 8 + (threadIdx.x >> 5);  // 16384 tiles
  const int tm = tile >> 5, tn = tile & 31;              // 512 M-tiles x 32 N-tiles
  const int row0 = tm * 16;
  const size_t wbase = (size_t)tn * (EENC / 32);
  v8f acc = {};
  for (int kk = 0; kk < EENC / 32; ++kk) {
    v16bf a = load_A_bf16(encbf, EENC, row0, kk * 32, lane);
    v16bf b = load_B_pack(Wkpk, wbase + kk, lane);
    acc = __builtin_amdgcn_wmma_f32_16x16x32_bf16(false, a, false, b, (short)0, acc, false, false);
  }
  const int n = tn * 16 + (lane & 15), half = lane >> 4;
  const float bv = bk[n];
#pragma unroll
  for (int r = 0; r < 8; ++r)
    out[(size_t)(row0 + r + 8 * half) * H_ + n] = acc[r] + bv;
}

// ---------------------------------------------------------------------------
// Per-step GEMM: out (32 x N) = X (32 x K, bf16, ldx) @ W^T + bias
// One wave per 16-wide N tile; both M tiles share the fragment-packed B.
// ---------------------------------------------------------------------------
__global__ __launch_bounds__(256)
void gemm_m32(const __bf16* __restrict__ X, int ldx, int K,
              const __bf16* __restrict__ Wp,
              const float* __restrict__ bias,      // nullable
              float* __restrict__ out, long long ldo) {
  const int lane = threadIdx.x & 31;
  const int tn = blockIdx.x * 8 + (threadIdx.x >> 5);
  const int ksteps = K >> 5;
  const size_t wbase = (size_t)tn * ksteps;
  v8f acc0 = {}, acc1 = {};
  for (int kk = 0; kk < ksteps; ++kk) {
    v16bf a0 = load_A_bf16(X, ldx, 0,  kk * 32, lane);
    v16bf a1 = load_A_bf16(X, ldx, 16, kk * 32, lane);
    v16bf bb = load_B_pack(Wp, wbase + kk, lane);
    acc0 = __builtin_amdgcn_wmma_f32_16x16x32_bf16(false, a0, false, bb, (short)0, acc0, false, false);
    acc1 = __builtin_amdgcn_wmma_f32_16x16x32_bf16(false, a1, false, bb, (short)0, acc1, false, false);
  }
  const int n = tn * 16 + (lane & 15), half = lane >> 4;
  const float bv = bias ? bias[n] : 0.0f;
#pragma unroll
  for (int r = 0; r < 8; ++r) {
    out[(size_t)(r + 8 * half) * ldo + n]      = acc0[r] + bv;
    out[(size_t)(16 + r + 8 * half) * ldo + n] = acc1[r] + bv;
  }
}

// ---------------------------------------------------------------------------
// Attention step: additive scores + tanh, softmax, context, emb gather.
// qproj is precomputed by a WMMA GEMM. One block per batch row, 256 threads.
// ---------------------------------------------------------------------------
__global__ __launch_bounds__(256)
void attn_step(const float* __restrict__ enc,    // (B,S,EENC) f32
               const float* __restrict__ kproj,  // (B*S,H)
               const float* __restrict__ qp,     // (B,H) = h1 @ Wq^T + bq
               const float* __restrict__ vatt,   // (H)
               const int*   __restrict__ tok,    // (B,T)
               const float* __restrict__ emb,    // (V,EEMB)
               __bf16* __restrict__ x0bf,        // (B,1280) [emb|ctx|h0]
               int t) {
  __shared__ float sqp[H_];
  __shared__ float ssc[S_];
  __shared__ float sred[S_];
  const int b = blockIdx.x, tid = threadIdx.x;

  sqp[tid]       = qp[b * H_ + tid];
  sqp[tid + 256] = qp[b * H_ + tid + 256];
  __syncthreads();

  // scores[s] = v . tanh(key_proj[b,s,:] + qproj) ; one wave per 32 s-values
  const int lane = tid & 31, wave = tid >> 5;
  for (int i = 0; i < 32; ++i) {
    int s = wave * 32 + i;
    const float* kp = kproj + ((size_t)b * S_ + s) * H_;
    float acc = 0.0f;
    for (int h = lane; h < H_; h += 32)
      acc += vatt[h] * tanhf(kp[h] + sqp[h]);
#pragma unroll
    for (int m = 16; m; m >>= 1) acc += __shfl_xor(acc, m, 32);
    if (lane == 0) ssc[s] = acc;
  }
  __syncthreads();

  // softmax over S=256 (one score per thread)
  float sc = ssc[tid];
  sred[tid] = sc; __syncthreads();
  for (int st = 128; st; st >>= 1) { if (tid < st) sred[tid] = fmaxf(sred[tid], sred[tid + st]); __syncthreads(); }
  float mx = sred[0]; __syncthreads();
  float ex = __expf(sc - mx);
  sred[tid] = ex; __syncthreads();
  for (int st = 128; st; st >>= 1) { if (tid < st) sred[tid] += sred[tid + st]; __syncthreads(); }
  float inv = 1.0f / sred[0]; __syncthreads();
  ssc[tid] = ex * inv;
  __syncthreads();

  // context[e] = sum_s w[s] * enc[b,s,e]  (coalesced over e)
#pragma unroll
  for (int jj = 0; jj < 2; ++jj) {
    int e = tid + jj * 256;
    const float* ep = enc + (size_t)b * S_ * EENC + e;
    float acc = 0.0f;
    for (int s = 0; s < S_; ++s) acc += ssc[s] * ep[(size_t)s * EENC];
    x0bf[(size_t)b * 1280 + EEMB + e] = f2bf(acc);
  }

  // embedding gather for this step
  int token = tok[b * T_ + t];
  x0bf[(size_t)b * 1280 + tid] = f2bf(emb[(size_t)token * EEMB + tid]);
}

// ---------------------------------------------------------------------------
// LSTM gate nonlinearities + state update + bf16 mirror maintenance.
// ---------------------------------------------------------------------------
__global__ __launch_bounds__(256)
void lstm_elem(const float* __restrict__ gates,   // (32,2048)
               const float* __restrict__ bih, const float* __restrict__ bhh,
               float* __restrict__ hbuf, float* __restrict__ cbuf,  // (32,512)
               int layer,
               __bf16* __restrict__ x0bf,         // layer0: h0 slot [:,768:1280]
               __bf16* __restrict__ xh1bf,        // layer0: [h0new | h1old]
               const float* __restrict__ h1old,   // layer0
               __bf16* __restrict__ xlogbf) {     // layer1: [h1new | ctx]
  const int idx = blockIdx.x * 256 + threadIdx.x; // 0..16383
  const int m = idx >> 9, j = idx & 511;
  const float* g = gates + (size_t)m * 2048;
  float gi = g[j]          + bih[j]          + bhh[j];
  float gf = g[512 + j]    + bih[512 + j]    + bhh[512 + j];
  float gg = g[1024 + j]   + bih[1024 + j]   + bhh[1024 + j];
  float go = g[1536 + j]   + bih[1536 + j]   + bhh[1536 + j];
  float c_new = sigm(gf) * cbuf[idx] + sigm(gi) * tanhf(gg);
  float h_new = sigm(go) * tanhf(c_new);
  cbuf[idx] = c_new;
  hbuf[idx] = h_new;
  __bf16 hb = f2bf(h_new);
  if (layer == 0) {
    x0bf[(size_t)m * 1280 + 768 + j]  = hb;
    xh1bf[(size_t)m * 1024 + j]       = hb;
    xh1bf[(size_t)m * 1024 + 512 + j] = f2bf(h1old[idx]);
  } else {
    xlogbf[(size_t)m * 1024 + j]       = hb;
    xlogbf[(size_t)m * 1024 + 512 + j] = x0bf[(size_t)m * 1280 + EEMB + j];
  }
}

// ---------------------------------------------------------------------------
extern "C" void kernel_launch(void* const* d_in, const int* in_sizes, int n_in,
                              void* d_out, int out_size, void* d_ws, size_t ws_size,
                              hipStream_t stream) {
  const float* enc    = (const float*)d_in[0];
  const int*   tok    = (const int*)  d_in[1];
  const float* emb    = (const float*)d_in[2];
  const float* Wq     = (const float*)d_in[3];
  const float* bq     = (const float*)d_in[4];
  const float* Wk     = (const float*)d_in[5];
  const float* bk     = (const float*)d_in[6];
  const float* vatt   = (const float*)d_in[7];
  const float* Wih0   = (const float*)d_in[8];
  const float* Whh0   = (const float*)d_in[9];
  const float* bih0   = (const float*)d_in[10];
  const float* bhh0   = (const float*)d_in[11];
  const float* Wih1   = (const float*)d_in[12];
  const float* Whh1   = (const float*)d_in[13];
  const float* bih1   = (const float*)d_in[14];
  const float* bhh1   = (const float*)d_in[15];
  const float* Wout   = (const float*)d_in[16];
  const float* bout   = (const float*)d_in[17];
  float* out = (float*)d_out;

  // -------- workspace layout (f32 region then bf16 region) ----------------
  float* h0    = (float*)d_ws;                 // 32*512
  float* h1    = h0 + 32 * 512;
  float* c0    = h1 + 32 * 512;
  float* c1    = c0 + 32 * 512;
  float* qproj = c1 + 32 * 512;                // 32*512
  float* gates = qproj + 32 * 512;             // 32*2048
  float* kproj = gates + 32 * 2048;            // 8192*512
  __bf16* encbf  = (__bf16*)(kproj + 8192 * 512);      // 8192*512
  __bf16* Wkpk   = encbf + (size_t)8192 * 512;         // 512*512   (frag order)
  __bf16* Wqpk   = Wkpk + 512 * 512;                   // 512*512   (frag order)
  __bf16* W0pk   = Wqpk + 512 * 512;                   // 2048*1280 [Wih0|Whh0]
  __bf16* W1pk   = W0pk + (size_t)2048 * 1280;         // 2048*1024 [Wih1|Whh1]
  __bf16* Woutpk = W1pk + (size_t)2048 * 1024;         // 32000*1024
  __bf16* x0bf   = Woutpk + (size_t)32000 * 1024;      // 32*1280 [emb|ctx|h0]
  __bf16* xh1bf  = x0bf + 32 * 1280;                   // 32*1024 [h0new|h1old]
  __bf16* xlogbf = xh1bf + 32 * 1024;                  // 32*1024 [h1new|ctx]

  // -------- init state: h=c=0, h0 slot of x0bf, xlogbf (qproj A at t=0) ----
  hipMemsetAsync(h0, 0, (size_t)4 * 32 * 512 * sizeof(float), stream);
  hipMemsetAsync(x0bf, 0, (size_t)32 * 1280 * sizeof(__bf16), stream);
  hipMemsetAsync(xlogbf, 0, (size_t)32 * 1024 * sizeof(__bf16), stream);

  // -------- one-time conversions / fragment packing ------------------------
  cvt_bf16_k<<<1024, 256, 0, stream>>>(enc, encbf, 8192 * 512);
  pack_frag_k<<<256, 256, 0, stream>>>(Wk, 512, nullptr, 0, Wkpk, 512, 512);
  pack_frag_k<<<256, 256, 0, stream>>>(Wq, 512, nullptr, 0, Wqpk, 512, 512);
  pack_frag_k<<<1024, 256, 0, stream>>>(Wih0, 768, Whh0, 512, W0pk, 2048, 1280);
  pack_frag_k<<<1024, 256, 0, stream>>>(Wih1, 512, Whh1, 512, W1pk, 2048, 1024);
  pack_frag_k<<<2048, 256, 0, stream>>>(Wout, 1024, nullptr, 0, Woutpk, 32000, 1024);

  // -------- key projection hoisted out of time loop ------------------------
  gemm_keyproj<<<2048, 256, 0, stream>>>(encbf, Wkpk, bk, kproj);

  // -------- decode loop ----------------------------------------------------
  for (int t = 0; t < T_; ++t) {
    // qproj = h1 @ Wq^T + bq  (A = bf16 h1 mirror in xlogbf[:,0:512])
    gemm_m32<<<4, 256, 0, stream>>>(xlogbf, 1024, 512, Wqpk, bq, qproj, 512LL);

    attn_step<<<B_, 256, 0, stream>>>(enc, kproj, qproj, vatt, tok, emb, x0bf, t);

    // layer 0: gates = [emb|ctx|h0] @ [Wih0|Whh0]^T   (N=2048, K=1280)
    gemm_m32<<<16, 256, 0, stream>>>(x0bf, 1280, 1280, W0pk, nullptr, gates, 2048LL);
    lstm_elem<<<64, 256, 0, stream>>>(gates, bih0, bhh0, h0, c0, 0, x0bf, xh1bf, h1, nullptr);

    // layer 1: gates = [h0new|h1old] @ [Wih1|Whh1]^T  (N=2048, K=1024)
    gemm_m32<<<16, 256, 0, stream>>>(xh1bf, 1024, 1024, W1pk, nullptr, gates, 2048LL);
    lstm_elem<<<64, 256, 0, stream>>>(gates, bih1, bhh1, h1, c1, 1, x0bf, nullptr, nullptr, xlogbf);

    // logits = [h1|ctx] @ Wout^T + bout -> out[:, t, :]  (N=32000, K=1024)
    gemm_m32<<<250, 256, 0, stream>>>(xlogbf, 1024, 1024, Woutpk, bout,
                                      out + (size_t)t * V_, (long long)T_ * V_);
  }

  // -------- final h, c outputs: [h0,h1,c0,c1] contiguous in ws -------------
  copy_f32_k<<<64, 256, 0, stream>>>(out + (size_t)B_ * T_ * V_, h0, 4 * 32 * 512);
}
